// Sparse_MHAT_16784732193199
// MI455X (gfx1250) — matrix operand order repaired and checked by reading, not compiled, not run on hardware
//
#include <hip/hip_runtime.h>
#include <cstddef>
#include <cstdint>

// ---------------------------------------------------------------------------
// Sparse strided MHA for MI455X (gfx1250, wave32, WMMA bf16, fp32 accum).
// B=4, L=4096, D=1024, H=16 (dk=64), M=4 slices (n=1024 tokens each).
//
// Every WMMA operand is stored PRE-PACKED in the ISA's wave32 fragment
// layout (CDNA5 ISA), so each fragment load is one contiguous
// 32-byte (2 x global_load_b128) access instead of 16 scalar u16 loads.
//
//   A-fragment (16x32 bf16): lane = m + 16*half; elem e -> k_local =
//        (e<8) ? e + 8*half : e + 8 + 8*half
//   B-fragment (32x16 bf16): lane = n + 16*half; elem e -> k_local as for A (V-pack and P: 16*half+e)
//   C/D (16x16 f32): VGPR r holds row r + 8*half, lane%16 = column
// ---------------------------------------------------------------------------

typedef __attribute__((ext_vector_type(16))) __bf16 v16bf;
typedef __attribute__((ext_vector_type(8)))  float  v8f;

constexpr int B_ = 4, L_ = 4096, D_ = 1024, H_ = 16, M_ = 4;
constexpr int DK    = D_ / H_;    // 64
constexpr int NTOK  = L_ / M_;    // 1024 tokens per slice
constexpr int FRAG  = 512;        // bf16 elements per packed fragment (32*16)
constexpr int KCH   = D_ / 32;    // 32 K-chunks in a D-length dot product

// ------------------- pack x (fp32, row-major) -> A-fragment bf16 -----------
__global__ void pack_x_bf16(const float* __restrict__ x,
                            __bf16* __restrict__ xp, size_t n) {
  size_t i      = (size_t)blockIdx.x * blockDim.x + threadIdx.x;
  size_t stride = (size_t)gridDim.x * blockDim.x;
  for (; i < n; i += stride) {
    const int e     = (int)(i & 15);
    const int lane  = (int)((i >> 4) & 31);
    const int chunk = (int)((i >> 9) & 31);
    const size_t tile = i >> 14;                 // 16-row tile
    const int ml = lane & 15, half = lane >> 4;
    const int kl = (e < 8) ? (e + 8 * half) : (e + 8 + 8 * half);
    xp[i] = (__bf16)x[(tile * 16 + ml) * D_ + chunk * 32 + kl];
  }
}

// ------------------- pack W (fp32 [K,N] row-major) -> B-fragment bf16 ------
__global__ void pack_w_bf16(const float* __restrict__ w,
                            __bf16* __restrict__ wp, size_t n) {
  size_t i      = (size_t)blockIdx.x * blockDim.x + threadIdx.x;
  size_t stride = (size_t)gridDim.x * blockDim.x;
  for (; i < n; i += stride) {
    const int e     = (int)(i & 15);
    const int lane  = (int)((i >> 4) & 31);
    const int ntile = (int)((i >> 9) & 63);
    const size_t ktile = i >> 15;
    const int nn = lane & 15, half = lane >> 4;
    const int kl = (e < 8) ? (e + 8 * half) : (e + 8 + 8 * half);
    wp[i] = (__bf16)w[(ktile * 32 + kl) * D_ + ntile * 16 + nn];
  }
}

// ---------------------------------------------------------------------------
// GEMM on packed operands. Block = 8 waves; wave computes a 16x64 strip.
// mode: 0 -> write Q-pack, 1 -> K-pack, 2 -> V-pack, 3 -> fp32 row-major out.
// ---------------------------------------------------------------------------
__global__ void gemm_pack(const __bf16* __restrict__ Ap,
                          const __bf16* __restrict__ Wp,
                          const float*  __restrict__ bias,
                          __bf16* __restrict__ outP,
                          float*  __restrict__ outF, int mode) {
  const int wave = threadIdx.x >> 5;
  const int lane = threadIdx.x & 31;
  const int ml   = lane & 15, half = lane >> 4;
  const int rt   = blockIdx.x;                       // 16-row tile index
  const int c0   = (blockIdx.y * 8 + wave) * 64;     // first output column

  v8f acc[4];
#pragma unroll
  for (int t = 0; t < 4; ++t) acc[t] = (v8f){0.f,0.f,0.f,0.f,0.f,0.f,0.f,0.f};

  const __bf16* abase = Ap + (size_t)rt * KCH * FRAG + lane * 16;
  const __bf16* wbase = Wp + lane * 16;

  for (int kc = 0; kc < KCH; ++kc) {
    const v16bf a = *(const v16bf*)(abase + (size_t)kc * FRAG);
#pragma unroll
    for (int t = 0; t < 4; ++t) {
      const v16bf bm =
          *(const v16bf*)(wbase + ((size_t)kc * (D_ / 16) + (c0 >> 4) + t) * FRAG);
      acc[t] = __builtin_amdgcn_wmma_f32_16x16x32_bf16(
          false, a, false, bm, (short)0, acc[t], false, false);
    }
  }

  if (mode == 3) {  // plain fp32 row-major output + bias
#pragma unroll
    for (int t = 0; t < 4; ++t) {
      const int col = c0 + t * 16 + ml;
      const float bv = bias[col];
#pragma unroll
      for (int r = 0; r < 8; ++r)
        outF[(size_t)(rt * 16 + r + 8 * half) * D_ + col] = acc[t][r] + bv;
    }
    return;
  }

  // scatter into the attention kernel's fragment layouts
#pragma unroll
  for (int t = 0; t < 4; ++t) {
    const int col = c0 + t * 16 + ml;
    const float bv = bias[col];
    const int h = col >> 6;      // head
    const int d = col & 63;      // dk coordinate
#pragma unroll
    for (int r = 0; r < 8; ++r) {
      const float val = acc[t][r] + bv;
      const int row = rt * 16 + r + 8 * half;   // global token row
      const int b  = row >> 12;                 // / L_
      const int tt = row & (L_ - 1);
      const int m  = tt & 3;
      const int j  = tt >> 2;                   // index within slice
      const size_t slice = ((size_t)b * H_ + h) * M_ + m;
      size_t idx;
      if (mode == 0) {            // Q: A-fragment tiles [qtile][chunk]
        const int qtile = j >> 4, m_l = j & 15;
        const int chunk = d >> 5, kl = d & 31;
        const int h2 = (kl >> 3) & 1;
        const int e  = kl - 8 * h2 - ((kl >= 16) ? 8 : 0);
        idx = ((slice * (NTOK / 16) + qtile) * 2 + chunk) * FRAG
              + (m_l + 16 * h2) * 16 + e;
      } else if (mode == 1) {     // K: B-fragment of K^T [ktile][chunk]
        const int ktile = j >> 4, jl = j & 15;
        const int chunk = d >> 5, kl = d & 31;
        const int h2 = (kl >> 3) & 1;
        const int e  = kl - 8 * h2 - ((kl >= 16) ? 8 : 0);
        idx = ((slice * (NTOK / 16) + ktile) * 2 + chunk) * FRAG
              + (jl + 16 * h2) * 16 + e;
      } else {                    // V: B-fragment [step][dk-tile]
        const int step = j >> 5, kk = j & 31;
        const int tc = d >> 4, nn = d & 15;
        idx = ((slice * (NTOK / 32) + step) * 4 + tc) * FRAG
              + (nn + 16 * (kk >> 4)) * 16 + (kk & 15);
      }
      outP[idx] = (__bf16)val;
    }
  }
}

// ---------------------------------------------------------------------------
// Flash attention over one (b, h, m) strided slice; all operands pre-packed.
// Block = 8 waves; wave owns 16 query rows; key loop in steps of 32 keys:
// 4 WMMA (S = Q K^T) + online softmax + LDS P re-layout + 4 WMMA (O += P V).
// Output written directly in A-fragment layout for the final projection GEMM.
// ---------------------------------------------------------------------------
__global__ void attn_kernel(const __bf16* __restrict__ Qp,
                            const __bf16* __restrict__ Kp,
                            const __bf16* __restrict__ Vp,
                            __bf16* __restrict__ Op) {
  __shared__ float sP[8][16][32];   // per-wave 16x32 probability tile (16 KB)

  const int wave = threadIdx.x >> 5;
  const int lane = threadIdx.x & 31;
  const int ml   = lane & 15, half = lane >> 4;

  int idx = blockIdx.x;
  const int qt = idx & 7;  idx >>= 3;
  const int m  = idx & 3;  idx >>= 2;
  const int h  = idx & 15; idx >>= 4;
  const int b  = idx;

  const size_t slice = ((size_t)b * H_ + h) * M_ + m;
  const int qtile = qt * 8 + wave;                   // wave's 16-query tile

  // Q fragments (16 q x 64 dk = 2 A-fragments), single b128-pair loads
  v16bf qa[2];
#pragma unroll
  for (int c = 0; c < 2; ++c)
    qa[c] = *(const v16bf*)(Qp + ((slice * (NTOK / 16) + qtile) * 2 + c) * FRAG
                            + lane * 16);

  float mrow[8], lrow[8];
  v8f acc[4];
#pragma unroll
  for (int r = 0; r < 8; ++r) { mrow[r] = -3.0e38f; lrow[r] = 0.f; }
#pragma unroll
  for (int t = 0; t < 4; ++t) acc[t] = (v8f){0.f,0.f,0.f,0.f,0.f,0.f,0.f,0.f};

  for (int step = 0; step < NTOK / 32; ++step) {
    // ----- S = (Q K^T) * 1/sqrt(dk): two 16x16 score tiles ----------------
    v8f s[2];
#pragma unroll
    for (int kt = 0; kt < 2; ++kt) {
      v8f sc = (v8f){0.f,0.f,0.f,0.f,0.f,0.f,0.f,0.f};
      const int ktile = step * 2 + kt;
#pragma unroll
      for (int c = 0; c < 2; ++c) {
        const v16bf kb_ =
            *(const v16bf*)(Kp + ((slice * (NTOK / 16) + ktile) * 2 + c) * FRAG
                            + lane * 16);
        sc = __builtin_amdgcn_wmma_f32_16x16x32_bf16(
            false, qa[c], false, kb_, (short)0, sc, false, false);
      }
      s[kt] = sc;
    }

    // ----- online softmax over the 32 new keys ----------------------------
    float alpha[8];
#pragma unroll
    for (int r = 0; r < 8; ++r) {
      const float t0 = s[0][r] * 0.125f;   // 1/sqrt(64)
      const float t1 = s[1][r] * 0.125f;
      float mx = fmaxf(t0, t1);
      mx = fmaxf(mx, __shfl_xor(mx, 8));
      mx = fmaxf(mx, __shfl_xor(mx, 4));
      mx = fmaxf(mx, __shfl_xor(mx, 2));
      mx = fmaxf(mx, __shfl_xor(mx, 1));
      const float nm = fmaxf(mrow[r], mx);
      alpha[r] = __expf(mrow[r] - nm);
      mrow[r]  = nm;
      const float p0 = __expf(t0 - nm);
      const float p1 = __expf(t1 - nm);
      float rs = p0 + p1;
      rs += __shfl_xor(rs, 8);
      rs += __shfl_xor(rs, 4);
      rs += __shfl_xor(rs, 2);
      rs += __shfl_xor(rs, 1);
      lrow[r] = lrow[r] * alpha[r] + rs;
      sP[wave][r + 8 * half][ml]      = p0;   // plain [row][col] stash
      sP[wave][r + 8 * half][16 + ml] = p1;
    }
#pragma unroll
    for (int t = 0; t < 4; ++t)
#pragma unroll
      for (int r = 0; r < 8; ++r) acc[t][r] *= alpha[r];

    // ----- re-layout P (C-layout -> A-fragment) through LDS ---------------
    v16bf pa;
#pragma unroll
    for (int e = 0; e < 16; e += 2) {
      const int v = e >> 1;
      const int kb = 16 * half + 2 * v;
      pa[e]     = (__bf16)sP[wave][ml][kb];
      pa[e + 1] = (__bf16)sP[wave][ml][kb + 1];
    }

    // ----- O += P @ V: 4 dk tiles -----------------------------------------
#pragma unroll
    for (int t = 0; t < 4; ++t) {
      const v16bf vb =
          *(const v16bf*)(Vp + ((slice * (NTOK / 32) + step) * 4 + t) * FRAG
                          + lane * 16);
      acc[t] = __builtin_amdgcn_wmma_f32_16x16x32_bf16(
          false, pa, false, vb, (short)0, acc[t], false, false);
    }
  }

  // ----- normalize; write directly in A-fragment layout for final GEMM ----
#pragma unroll
  for (int t = 0; t < 4; ++t) {
#pragma unroll
    for (int r = 0; r < 8; ++r) {
      const float o = acc[t][r] / lrow[r];
      const int q    = qtile * 16 + r + 8 * half;
      const int trow = b * L_ + m + q * M_;            // global token row
      const int col  = h * 64 + t * 16 + ml;           // feature column
      const int tile = trow >> 4, m_l2 = trow & 15;
      const int chunk = col >> 5, kl = col & 31;
      const int h2 = (kl >> 3) & 1;
      const int e  = kl - 8 * h2 - ((kl >= 16) ? 8 : 0);
      Op[((size_t)tile * KCH + chunk) * FRAG + (m_l2 + 16 * h2) * 16 + e] =
          (__bf16)o;
    }
  }
}

// ---------------------------------------------------------------------------
extern "C" void kernel_launch(void* const* d_in, const int* in_sizes, int n_in,
                              void* d_out, int out_size, void* d_ws, size_t ws_size,
                              hipStream_t stream) {
  (void)in_sizes; (void)n_in; (void)out_size; (void)ws_size;

  const float* x  = (const float*)d_in[0];
  const float* Wq = (const float*)d_in[1];
  const float* bq = (const float*)d_in[2];
  const float* Wk = (const float*)d_in[3];
  const float* bk = (const float*)d_in[4];
  const float* Wv = (const float*)d_in[5];
  const float* bv = (const float*)d_in[6];
  const float* Wo = (const float*)d_in[7];
  const float* bo = (const float*)d_in[8];
  float* out = (float*)d_out;

  const size_t NT = (size_t)B_ * L_;   // 16384 token rows
  const size_t ME = NT * D_;           // activation elements (16.7M)
  const size_t WE = (size_t)D_ * D_;   // weight elements (1M)

  __bf16* ws  = (__bf16*)d_ws;         // 5*ME + 4*WE bf16 ~ 176 MB (fits L2)
  __bf16* Xp  = ws;
  __bf16* Qp  = ws + ME;
  __bf16* Kp  = ws + 2 * ME;
  __bf16* Vp  = ws + 3 * ME;
  __bf16* Op  = ws + 4 * ME;
  __bf16* Wqp = ws + 5 * ME;
  __bf16* Wkp = Wqp + WE;
  __bf16* Wvp = Wkp + WE;
  __bf16* Wop = Wvp + WE;

  // 1) convert + pack into WMMA fragment layouts
  pack_x_bf16<<<2048, 256, 0, stream>>>(x, Xp, ME);
  pack_w_bf16<<<512,  256, 0, stream>>>(Wq, Wqp, WE);
  pack_w_bf16<<<512,  256, 0, stream>>>(Wk, Wkp, WE);
  pack_w_bf16<<<512,  256, 0, stream>>>(Wv, Wvp, WE);
  pack_w_bf16<<<512,  256, 0, stream>>>(Wo, Wop, WE);

  // 2) Q/K/V projections, epilogue scatters into attention fragment layouts
  dim3 gGemm(NT / 16, D_ / (8 * 64));  // (1024, 2)
  gemm_pack<<<gGemm, 256, 0, stream>>>(Xp, Wqp, bq, Qp, nullptr, 0);
  gemm_pack<<<gGemm, 256, 0, stream>>>(Xp, Wkp, bk, Kp, nullptr, 1);
  gemm_pack<<<gGemm, 256, 0, stream>>>(Xp, Wvp, bv, Vp, nullptr, 2);

  // 3) strided-slice flash attention (B*H*M slices x 8 query tiles)
  attn_kernel<<<B_ * H_ * M_ * (NTOK / 128), 256, 0, stream>>>(Qp, Kp, Vp, Op);

  // 4) output projection -> fp32 + bias
  gemm_pack<<<gGemm, 256, 0, stream>>>(Op, Wop, bo, nullptr, out, 3);
}
